// TSAEAdjacentContrastive_22016002359839
// MI455X (gfx1250) — matrix operand order repaired and checked by reading, not compiled, not run on hardware
//
#include <hip/hip_runtime.h>

#define DIN  768
#define DSAE 24576
#define NTOK 8192
#define TOPK 64

#define BM 128
#define BN 128
#define BK 32
#define LDK 40              // LDS k-stride (bf16 elems): 80B rows, 16B aligned
#define KSTEPS (DIN / BK)   // 24

typedef __attribute__((ext_vector_type(16))) __bf16        v16bf;
typedef __attribute__((ext_vector_type(8)))  float         v8f;
typedef __attribute__((ext_vector_type(4)))  float         f32x4;
typedef __attribute__((ext_vector_type(4)))  unsigned int  u32x4;
typedef __attribute__((ext_vector_type(2)))  unsigned int  u32x2;

// round-to-nearest-even f32->bf16, two at a time, packed into one dword
__device__ __forceinline__ unsigned pack_bf16x2(float lo, float hi) {
  unsigned ul = __float_as_uint(lo);
  unsigned uh = __float_as_uint(hi);
  ul += 0x7FFFu + ((ul >> 16) & 1u);
  uh += 0x7FFFu + ((uh >> 16) & 1u);
  return (ul >> 16) | (uh & 0xFFFF0000u);
}

// order-preserving map: f32 -> u32 (ascending)
__device__ __forceinline__ unsigned f32_key(float f) {
  unsigned u = __float_as_uint(f);
  return u ^ ((u & 0x80000000u) ? 0xFFFFFFFFu : 0x80000000u);
}

// -----------------------------------------------------------------------------
// Kernel 1: pre = (x - b_dec) @ W_enc + b_enc   (bf16 WMMA, f32 accumulate)
// double-buffered LDS, register-staged global loads, NT stores for pre.
// -----------------------------------------------------------------------------
__global__ __launch_bounds__(256)
void sae_encode_wmma(const float* __restrict__ x,
                     const float* __restrict__ Wenc,
                     const float* __restrict__ benc,
                     const float* __restrict__ bdec,
                     float* __restrict__ pre) {
  __shared__ unsigned short As[2][BM][LDK];   // [m][k]
  __shared__ unsigned short Bs[2][BN][LDK];   // [n][k] (transposed at store)

  const int tid  = threadIdx.x;
  const int M0   = blockIdx.y * BM;
  const int N0   = blockIdx.x * BN;
  const int lane = tid & 31;
  const int wv   = tid >> 5;      // 8 waves
  const int wm   = wv & 3;        // 4 row-slabs of 32
  const int wn   = wv >> 2;       // 2 col-slabs of 64
  const int g    = lane >> 4;
  const int l16  = lane & 15;

  f32x4 ra[4], rbd[4], rb[4];

  auto load_chunk = [&](int kb) {
    #pragma unroll
    for (int i = 0; i < 4; ++i) {             // A: 128 rows x 32 k
      int idx = tid + i * 256;                // 0..1023
      int r   = idx >> 3;                     // 0..127
      int kq  = (idx & 7) << 2;               // 0,4,...,28
      ra[i]  = *(const f32x4*)(x + (size_t)(M0 + r) * DIN + kb + kq);
      rbd[i] = *(const f32x4*)(bdec + kb + kq);
    }
    #pragma unroll
    for (int i = 0; i < 2; ++i) {             // B: 32 k x 128 n, 2 k-rows/thread
      int idx = tid + i * 256;                // 0..511
      int r   = (idx >> 5) << 1;              // 0,2,...,30
      int c   = (idx & 31) << 2;              // 0,4,...,124
      rb[2*i]   = *(const f32x4*)(Wenc + (size_t)(kb + r)     * DSAE + N0 + c);
      rb[2*i+1] = *(const f32x4*)(Wenc + (size_t)(kb + r + 1) * DSAE + N0 + c);
    }
  };

  auto store_chunk = [&](int buf) {
    #pragma unroll
    for (int i = 0; i < 4; ++i) {             // A tile: one b64 store per float4
      int idx = tid + i * 256;
      int r   = idx >> 3;
      int kq  = (idx & 7) << 2;
      u32x2 p;
      p.x = pack_bf16x2(ra[i].x - rbd[i].x, ra[i].y - rbd[i].y);
      p.y = pack_bf16x2(ra[i].z - rbd[i].z, ra[i].w - rbd[i].w);
      *(u32x2*)&As[buf][r][kq] = p;
    }
    #pragma unroll
    for (int i = 0; i < 2; ++i) {             // B tile: b32 stores ({k,k+1} pair)
      int idx = tid + i * 256;
      int r   = (idx >> 5) << 1;
      int c   = (idx & 31) << 2;
      f32x4 lo = rb[2*i], hi = rb[2*i+1];
      *(unsigned*)&Bs[buf][c + 0][r] = pack_bf16x2(lo.x, hi.x);
      *(unsigned*)&Bs[buf][c + 1][r] = pack_bf16x2(lo.y, hi.y);
      *(unsigned*)&Bs[buf][c + 2][r] = pack_bf16x2(lo.z, hi.z);
      *(unsigned*)&Bs[buf][c + 3][r] = pack_bf16x2(lo.w, hi.w);
    }
  };

  v8f acc[2][4];
  #pragma unroll
  for (int i = 0; i < 2; ++i)
    #pragma unroll
    for (int j = 0; j < 4; ++j)
      acc[i][j] = (v8f){0.f, 0.f, 0.f, 0.f, 0.f, 0.f, 0.f, 0.f};

  load_chunk(0);
  store_chunk(0);
  __syncthreads();

  int buf = 0;
  for (int t = 0; t < KSTEPS; ++t) {
    const bool more = (t + 1) < KSTEPS;
    if (more) load_chunk((t + 1) * BK);       // global loads overlap the WMMAs

    union Frag { v16bf v; u32x4 q[2]; } fa[2], fb[4];
    #pragma unroll
    for (int sm = 0; sm < 2; ++sm) {
      const unsigned short* p = &As[buf][wm * 32 + sm * 16 + l16][0];
      fa[sm].q[0] = *(const u32x4*)(p + g * 8);        // K = g*8 + 0..7
      fa[sm].q[1] = *(const u32x4*)(p + 16 + g * 8);   // K = 16 + g*8 + 0..7
    }
    #pragma unroll
    for (int sn = 0; sn < 4; ++sn) {
      const unsigned short* p = &Bs[buf][wn * 64 + sn * 16 + l16][0];
      fb[sn].q[0] = *(const u32x4*)(p + g * 8);        // K = g*8 + 0..7
      fb[sn].q[1] = *(const u32x4*)(p + 16 + g * 8);   // K = 16 + g*8 + 0..7
    }
    #pragma unroll
    for (int sm = 0; sm < 2; ++sm)
      #pragma unroll
      for (int sn = 0; sn < 4; ++sn)
        acc[sm][sn] = __builtin_amdgcn_wmma_f32_16x16x32_bf16(
            false, fa[sm].v, false, fb[sn].v, (short)0, acc[sm][sn],
            false, false);

    if (more) store_chunk(buf ^ 1);
    __syncthreads();
    buf ^= 1;
  }

  // epilogue: + b_enc, NT-store pre (C layout: row = r + 8*g, col = l16)
  #pragma unroll
  for (int sn = 0; sn < 4; ++sn) {
    const int col  = N0 + wn * 64 + sn * 16 + l16;
    const float be = benc[col];
    #pragma unroll
    for (int sm = 0; sm < 2; ++sm) {
      const int rbase = M0 + wm * 32 + sm * 16 + g * 8;
      #pragma unroll
      for (int r = 0; r < 8; ++r)
        __builtin_nontemporal_store(acc[sm][sn][r] + be,
                                    pre + (size_t)(rbase + r) * DSAE + col);
    }
  }
}

// -----------------------------------------------------------------------------
// Kernel 2: per-row exact top-64 (bitwise radix descent on float keys),
// mask the row in LDS, stream it back NT-vectorized, sparse decode x_hat.
// -----------------------------------------------------------------------------
__global__ __launch_bounds__(256)
void sae_topk_decode(const float* __restrict__ Wdec,
                     const float* __restrict__ bdec,
                     float* __restrict__ out) {
  const int row = blockIdx.x;
  float* zrow = out + (size_t)NTOK * DIN + (size_t)row * DSAE;
  float* xrow = out + (size_t)row * DIN;

  __shared__ float    s_vals[DSAE];     // 96 KB row stage (CDNA5: 320KB/WG)
  __shared__ int      s_red[8];
  __shared__ unsigned s_prefix;
  __shared__ int      s_krem;
  __shared__ int      s_tie, s_nsel;
  __shared__ int      s_selIdx[TOPK];
  __shared__ float    s_selVal[TOPK];
  __shared__ int      s_srtIdx[TOPK];
  __shared__ float    s_srtVal[TOPK];

  const int tid  = threadIdx.x;
  const int lane = tid & 31;
  const int wid  = tid >> 5;

  // stage row into LDS (NT: read-once stream)
  for (int i = tid; i < DSAE / 4; i += 256)
    ((f32x4*)s_vals)[i] = __builtin_nontemporal_load(((const f32x4*)zrow) + i);
  if (tid == 0) { s_prefix = 0u; s_krem = TOPK; s_tie = 0; s_nsel = 0; }
  __syncthreads();

  // 32-step bitwise radix descent: exact key of the K-th largest element
  for (int bit = 31; bit >= 0; --bit) {
    const unsigned test = s_prefix | (1u << bit);
    int c = 0;
    for (int i = tid; i < DSAE; i += 256) {
      unsigned key = f32_key(s_vals[i]);
      c += (int)(((key ^ test) >> bit) == 0u);
    }
    #pragma unroll
    for (int off = 16; off > 0; off >>= 1) c += __shfl_down(c, off, 32);
    if (lane == 0) s_red[wid] = c;
    __syncthreads();
    if (tid == 0) {
      int tot = 0;
      #pragma unroll
      for (int i = 0; i < 8; ++i) tot += s_red[i];
      if (tot >= s_krem) s_prefix |= (1u << bit);
      else               s_krem   -= tot;
    }
    __syncthreads();
  }
  const unsigned T    = s_prefix;
  const int      need = s_krem;
  __syncthreads();

  // select winners; zero the rejects in LDS (owner-thread writes only)
  for (int i = tid; i < DSAE; i += 256) {
    float v = s_vals[i];
    unsigned key = f32_key(v);
    bool keep = false;
    if (key > T) {
      keep = true;
    } else if (key == T) {
      if (atomicAdd(&s_tie, 1) < need) keep = true;
    }
    if (keep) {
      int s = atomicAdd(&s_nsel, 1);
      s_selIdx[s] = i;
      s_selVal[s] = v;
    } else {
      s_vals[i] = 0.0f;
    }
  }
  __syncthreads();

  // stream masked row back to z, vectorized + non-temporal
  for (int i = tid; i < DSAE / 4; i += 256)
    __builtin_nontemporal_store(((const f32x4*)s_vals)[i], ((f32x4*)zrow) + i);

  // rank-sort survivors by index (unique) -> deterministic summation order
  if (tid < TOPK) {
    int   mi = s_selIdx[tid];
    float mv = s_selVal[tid];
    int rank = 0;
    #pragma unroll 8
    for (int s = 0; s < TOPK; ++s) rank += (int)(s_selIdx[s] < mi);
    s_srtIdx[rank] = mi;
    s_srtVal[rank] = mv;
  }
  __syncthreads();

  // sparse decode: x_hat[row, c] = b_dec[c] + sum_s val_s * W_dec[idx_s, c]
  for (int c = tid; c < DIN; c += 256) {
    float acc = bdec[c];
    #pragma unroll 8
    for (int s = 0; s < TOPK; ++s)
      acc += s_srtVal[s] * Wdec[(size_t)s_srtIdx[s] * DIN + c];
    xrow[c] = acc;
  }
}

// -----------------------------------------------------------------------------
extern "C" void kernel_launch(void* const* d_in, const int* in_sizes, int n_in,
                              void* d_out, int out_size, void* d_ws, size_t ws_size,
                              hipStream_t stream) {
  (void)in_sizes; (void)n_in; (void)out_size; (void)d_ws; (void)ws_size;
  const float* x    = (const float*)d_in[0];
  const float* Wenc = (const float*)d_in[1];
  const float* Wdec = (const float*)d_in[2];
  const float* benc = (const float*)d_in[3];
  const float* bdec = (const float*)d_in[4];
  float* out = (float*)d_out;
  float* pre = out + (size_t)NTOK * DIN;     // z region doubles as pre scratch

  dim3 g1(DSAE / BN, NTOK / BM);             // 192 x 64
  sae_encode_wmma<<<g1, 256, 0, stream>>>(x, Wenc, benc, bdec, pre);
  sae_topk_decode<<<NTOK, 256, 0, stream>>>(Wdec, bdec, out);
}
